// WaveRNN_37495064494155
// MI455X (gfx1250) — matrix operand order repaired and checked by reading, not compiled, not run on hardware
//
#include <hip/hip_runtime.h>
#include <hip/hip_bf16.h>

#define B_SZ 32768
#define H_SZ 896
#define S_SZ 448
#define Q_SZ 256
#define H3   (3 * H_SZ)   // 2688
#define S3   (3 * S_SZ)   // 1344

typedef __bf16 bf16;
typedef __bf16 v16bf __attribute__((ext_vector_type(16)));
typedef __bf16 v8bf  __attribute__((ext_vector_type(8)));
typedef float  v8f   __attribute__((ext_vector_type(8)));
typedef float  v4f   __attribute__((ext_vector_type(4)));

static __device__ __forceinline__ v16bf cat8(v8bf lo, v8bf hi) {
  return __builtin_shufflevector(lo, hi, 0,1,2,3,4,5,6,7,8,9,10,11,12,13,14,15);
}

// B fragment (32x16 bf16): lane n = lane&15, K-offset 0/8 by lane half, elems 8..15 at +16.
// Wt is column-major bf16: Wt[col][k], row length = K. Two contiguous 16B loads.
static __device__ __forceinline__ v16bf ldB(const bf16* __restrict__ Wt, int K,
                                            int col, int k0, int lane) {
  int n    = lane & 15;
  int koff = (lane >> 4) << 3;            // 0 or 8
  const bf16* p = Wt + (size_t)(col + n) * K + k0 + koff;
  return cat8(*(const v8bf*)p, *(const v8bf*)(p + 16));
}

static __device__ __forceinline__ float sigm(float x) {
  return 1.0f / (1.0f + __expf(-x));
}
static __device__ __forceinline__ float tanh_f(float x) {
  float cx = fminf(fmaxf(x, -15.f), 15.f);
  float e  = __expf(2.0f * cx);
  return (e - 1.0f) / (e + 1.0f);
}

#define WMMA_BF16(A, Bm, C) \
  __builtin_amdgcn_wmma_f32_16x16x32_bf16(false, (A), false, (Bm), (short)0, (C), false, false)

// ---------------------------------------------------------------------------
// Weight convert + transpose: src fp32 [K][N] row-major -> dst bf16 [N][K]
// ---------------------------------------------------------------------------
__global__ void transpose_bf16_kernel(const float* __restrict__ src,
                                      bf16* __restrict__ dst, int K, int N) {
  long id = (long)blockIdx.x * blockDim.x + threadIdx.x;
  long total = (long)K * N;
  if (id >= total) return;
  int nn = (int)(id / K);
  int kk = (int)(id % K);
  dst[id] = (bf16)src[(size_t)kk * N + nn];
}

// ---------------------------------------------------------------------------
// Fused GRU: R = prev_hidden @ W_R (bf16 WMMA), + rank-2/3 input projections,
// gates, blend. Tile: 64 rows x 64 cols per gate. 8 waves = 4(M) x 2(N).
// K-loop is explicitly double-buffered: iteration k issues loads for k+32,
// then runs 6 WMMAs on already-resident fragments (no load-wait before WMMA).
// ---------------------------------------------------------------------------
__global__ __launch_bounds__(256) void gru_kernel(
    const float* __restrict__ prev_y, const float* __restrict__ prev_hidden,
    const float* __restrict__ cur_c,  const bf16* __restrict__ Wrt,
    const float* __restrict__ W_Ic,   const float* __restrict__ W_If,
    const float* __restrict__ bias_u, const float* __restrict__ bias_r,
    const float* __restrict__ bias_e,
    float* __restrict__ hidden_out,   bf16* __restrict__ hidden_bf) {
  int lane = threadIdx.x & 31;
  int w    = threadIdx.x >> 5;
  int mw   = w >> 1, nw = w & 1;
  int rowBase = blockIdx.y * 64 + mw * 16;            // batch rows of this wave
  int colBase = blockIdx.x * 64 + nw * 32;            // per-gate column base

  v8f aU[2], aR[2], aE[2];
#pragma unroll
  for (int s = 0; s < 2; ++s)
#pragma unroll
    for (int i = 0; i < 8; ++i) { aU[s][i] = 0.f; aR[s][i] = 0.f; aE[s][i] = 0.f; }

  int m  = lane & 15;
  int kb = (lane >> 4) << 3;                          // 0 or 8
  const float* parow = prev_hidden + (size_t)(rowBase + m) * H_SZ + kb;

  auto loadA = [&](int k0) -> v16bf {
    v4f f0 = *(const v4f*)(parow + k0);
    v4f f1 = *(const v4f*)(parow + k0 + 4);
    v4f f2 = *(const v4f*)(parow + k0 + 16);
    v4f f3 = *(const v4f*)(parow + k0 + 20);
    v16bf a;
#pragma unroll
    for (int i = 0; i < 4; ++i) {
      a[i]      = (bf16)f0[i];
      a[4 + i]  = (bf16)f1[i];
      a[8 + i]  = (bf16)f2[i];
      a[12 + i] = (bf16)f3[i];
    }
    return a;
  };

  // prologue: fragments for k0 = 0
  v16bf a = loadA(0);
  v16bf b[6];
#pragma unroll
  for (int g = 0; g < 3; ++g)
#pragma unroll
    for (int s = 0; s < 2; ++s)
      b[g * 2 + s] = ldB(Wrt, H_SZ, g * H_SZ + colBase + s * 16, 0, lane);

  // steady state: prefetch (k0), compute (k0-32)
#pragma unroll 2
  for (int k0 = 32; k0 < H_SZ; k0 += 32) {
    v16bf an = loadA(k0);
    v16bf bn[6];
#pragma unroll
    for (int g = 0; g < 3; ++g)
#pragma unroll
      for (int s = 0; s < 2; ++s)
        bn[g * 2 + s] = ldB(Wrt, H_SZ, g * H_SZ + colBase + s * 16, k0, lane);
#pragma unroll
    for (int s = 0; s < 2; ++s) {
      aU[s] = WMMA_BF16(a, b[0 * 2 + s], aU[s]);
      aR[s] = WMMA_BF16(a, b[1 * 2 + s], aR[s]);
      aE[s] = WMMA_BF16(a, b[2 * 2 + s], aE[s]);
    }
    a = an;
#pragma unroll
    for (int i = 0; i < 6; ++i) b[i] = bn[i];
  }
  // epilogue of pipeline: last K step
#pragma unroll
  for (int s = 0; s < 2; ++s) {
    aU[s] = WMMA_BF16(a, b[0 * 2 + s], aU[s]);
    aR[s] = WMMA_BF16(a, b[1 * 2 + s], aR[s]);
    aE[s] = WMMA_BF16(a, b[2 * 2 + s], aE[s]);
  }

  // Epilogue: C layout -> lane n = lane&15; elem i -> row (i + 8*(lane>>4))
  int n   = lane & 15;
  int mhi = (lane >> 4) ? 8 : 0;
#pragma unroll
  for (int s = 0; s < 2; ++s) {
    int j = colBase + s * 16 + n;                     // hidden column [0,896)
    float bu = bias_u[j], br = bias_r[j], be = bias_e[j];
    bool coarse = (j < S_SZ);
    int  jj = coarse ? j : j - S_SZ;
    const float* WI = coarse ? W_Ic : W_If;
    float wu0 = WI[0 * S3 + jj],            wu1 = WI[1 * S3 + jj];
    float wr0 = WI[0 * S3 + S_SZ + jj],     wr1 = WI[1 * S3 + S_SZ + jj];
    float we0 = WI[0 * S3 + 2 * S_SZ + jj], we1 = WI[1 * S3 + 2 * S_SZ + jj];
    float wu2 = 0.f, wr2 = 0.f, we2 = 0.f;
    if (!coarse) {
      wu2 = W_If[2 * S3 + jj];
      wr2 = W_If[2 * S3 + S_SZ + jj];
      we2 = W_If[2 * S3 + 2 * S_SZ + jj];
    }
#pragma unroll
    for (int i = 0; i < 8; ++i) {
      int bi = rowBase + mhi + i;
      float y0 = prev_y[2 * bi], y1 = prev_y[2 * bi + 1];
      float c2 = coarse ? 0.f : cur_c[bi];
      float Iu = y0 * wu0 + y1 * wu1 + c2 * wu2;
      float Ir = y0 * wr0 + y1 * wr1 + c2 * wr2;
      float Ie = y0 * we0 + y1 * we1 + c2 * we2;
      float u  = sigm(aU[s][i] + Iu + bu);
      float r  = sigm(aR[s][i] + Ir + br);
      float e  = tanh_f(r * aE[s][i] + Ie + be);
      float ph = prev_hidden[(size_t)bi * H_SZ + j];
      float h  = u * ph + (1.0f - u) * e;
      hidden_out[(size_t)bi * H_SZ + j] = h;
      hidden_bf[(size_t)bi * H_SZ + j]  = (bf16)h;
    }
  }
}

// ---------------------------------------------------------------------------
// Fused output head: out = relu(h @ W1 + b1) @ W2 + b2.
// h: bf16 [B][896] (+colOff selects coarse/fine half, K=448).
// Phase 1 -> T (64x448 bf16) in LDS; phase 2 -> out (64x256 fp32).
// B-fragment loads are batched (7 / 8 at a time) ahead of their WMMAs so the
// loads clause together and the WMMA burst runs behind a single wait.
// ---------------------------------------------------------------------------
__global__ __launch_bounds__(256) void head_kernel(
    const bf16* __restrict__ hb, int colOff,
    const bf16* __restrict__ W1t, const float* __restrict__ b1,
    const bf16* __restrict__ W2t, const float* __restrict__ b2,
    float* __restrict__ out) {
  __shared__ bf16 Ts[64 * 456];                       // 456 stride keeps 16B align
  int lane = threadIdx.x & 31;
  int w    = threadIdx.x >> 5;
  int mw   = w >> 1, nw = w & 1;
  int rowBase = blockIdx.x * 64 + mw * 16;
  int m  = lane & 15;
  int kb = (lane >> 4) << 3;
  int n  = lane & 15;
  int mhi = (lane >> 4) ? 8 : 0;

  // ---- phase 1: T = relu(h @ W1 + b1); this wave: cols nw*224 + t*16 ----
  v8f acc[14];
#pragma unroll
  for (int t = 0; t < 14; ++t)
#pragma unroll
    for (int i = 0; i < 8; ++i) acc[t][i] = 0.f;

  const bf16* pa = hb + (size_t)(rowBase + m) * H_SZ + colOff + kb;
  for (int k0 = 0; k0 < S_SZ; k0 += 32) {
    v16bf a = cat8(*(const v8bf*)(pa + k0), *(const v8bf*)(pa + k0 + 16));
#pragma unroll
    for (int half = 0; half < 2; ++half) {
      v16bf bfr[7];
#pragma unroll
      for (int t = 0; t < 7; ++t)
        bfr[t] = ldB(W1t, S_SZ, nw * 224 + (half * 7 + t) * 16, k0, lane);
#pragma unroll
      for (int t = 0; t < 7; ++t)
        acc[half * 7 + t] = WMMA_BF16(a, bfr[t], acc[half * 7 + t]);
    }
  }
#pragma unroll
  for (int t = 0; t < 14; ++t) {
    int j = nw * 224 + t * 16 + n;
    float bb = b1[j];
#pragma unroll
    for (int i = 0; i < 8; ++i) {
      int r = mw * 16 + mhi + i;
      float v = acc[t][i] + bb;
      v = v > 0.f ? v : 0.f;
      Ts[r * 456 + j] = (bf16)v;
    }
  }
  __syncthreads();

  // ---- phase 2: out = T @ W2 + b2; this wave: cols nw*128 + t*16 ----
  v8f acc2[8];
#pragma unroll
  for (int t = 0; t < 8; ++t)
#pragma unroll
    for (int i = 0; i < 8; ++i) acc2[t][i] = 0.f;

  for (int k0 = 0; k0 < S_SZ; k0 += 32) {
    const bf16* pt = &Ts[(mw * 16 + m) * 456 + k0 + kb];
    v16bf a = cat8(*(const v8bf*)pt, *(const v8bf*)(pt + 16));
    v16bf bfr[8];
#pragma unroll
    for (int t = 0; t < 8; ++t)
      bfr[t] = ldB(W2t, S_SZ, nw * 128 + t * 16, k0, lane);
#pragma unroll
    for (int t = 0; t < 8; ++t)
      acc2[t] = WMMA_BF16(a, bfr[t], acc2[t]);
  }
#pragma unroll
  for (int t = 0; t < 8; ++t) {
    int q = nw * 128 + t * 16 + n;
    float bb = b2[q];
#pragma unroll
    for (int i = 0; i < 8; ++i) {
      int r = rowBase + mhi + i;
      out[(size_t)r * Q_SZ + q] = acc2[t][i] + bb;
    }
  }
}

// ---------------------------------------------------------------------------
extern "C" void kernel_launch(void* const* d_in, const int* in_sizes, int n_in,
                              void* d_out, int out_size, void* d_ws, size_t ws_size,
                              hipStream_t stream) {
  const float* prev_y = (const float*)d_in[0];
  const float* prev_h = (const float*)d_in[1];
  const float* cur_c  = (const float*)d_in[2];
  const float* W_R    = (const float*)d_in[3];
  const float* W_Ic   = (const float*)d_in[4];
  const float* W_If   = (const float*)d_in[5];
  const float* W_O1   = (const float*)d_in[6];
  const float* b_O1   = (const float*)d_in[7];
  const float* W_O2   = (const float*)d_in[8];
  const float* b_O2   = (const float*)d_in[9];
  const float* W_O3   = (const float*)d_in[10];
  const float* b_O3   = (const float*)d_in[11];
  const float* W_O4   = (const float*)d_in[12];
  const float* b_O4   = (const float*)d_in[13];
  const float* bias_u = (const float*)d_in[14];
  const float* bias_r = (const float*)d_in[15];
  const float* bias_e = (const float*)d_in[16];
  float* out = (float*)d_out;
  char*  ws  = (char*)d_ws;

  // ws layout (bytes)
  const size_t offWrt = 0;                                   // [2688][896] bf16
  const size_t offW1t = offWrt + (size_t)H3 * H_SZ * 2;      // [448][448]
  const size_t offW2t = offW1t + (size_t)S_SZ * S_SZ * 2;    // [256][448]
  const size_t offW3t = offW2t + (size_t)Q_SZ * S_SZ * 2;    // [448][448]
  const size_t offW4t = offW3t + (size_t)S_SZ * S_SZ * 2;    // [256][448]
  const size_t offHbf = offW4t + (size_t)Q_SZ * S_SZ * 2;    // [B][896] bf16
  bf16* Wrt = (bf16*)(ws + offWrt);
  bf16* W1t = (bf16*)(ws + offW1t);
  bf16* W2t = (bf16*)(ws + offW2t);
  bf16* W3t = (bf16*)(ws + offW3t);
  bf16* W4t = (bf16*)(ws + offW4t);
  bf16* hbf = (bf16*)(ws + offHbf);

  auto launchT = [&](const float* s, bf16* d, int K, int N) {
    long tot = (long)K * N;
    transpose_bf16_kernel<<<(unsigned)((tot + 255) / 256), 256, 0, stream>>>(s, d, K, N);
  };
  launchT(W_R,  Wrt, H_SZ, H3);
  launchT(W_O1, W1t, S_SZ, S_SZ);
  launchT(W_O2, W2t, S_SZ, Q_SZ);
  launchT(W_O3, W3t, S_SZ, S_SZ);
  launchT(W_O4, W4t, S_SZ, Q_SZ);

  float* hidden_out = out + 2 * (size_t)B_SZ * Q_SZ;
  gru_kernel<<<dim3(H_SZ / 64, B_SZ / 64), 256, 0, stream>>>(
      prev_y, prev_h, cur_c, Wrt, W_Ic, W_If, bias_u, bias_r, bias_e,
      hidden_out, hbf);

  head_kernel<<<B_SZ / 64, 256, 0, stream>>>(hbf, 0,    W1t, b_O1, W2t, b_O2, out);
  head_kernel<<<B_SZ / 64, 256, 0, stream>>>(hbf, S_SZ, W3t, b_O3, W4t, b_O4,
                                             out + (size_t)B_SZ * Q_SZ);
}
